// AttendCompareAggregateBlock_67611375173860
// MI455X (gfx1250) — matrix operand order repaired and checked by reading, not compiled, not run on hardware
//
#include <hip/hip_runtime.h>
#include <hip/hip_bf16.h>

typedef __attribute__((ext_vector_type(16))) _Float16 v16h;
typedef __attribute__((ext_vector_type(8)))  float    v8f;

// ---------------- problem constants ----------------
#define BB   4
#define CCH  512
#define TT   4
#define HP   31
#define WP   31
#define HWP  961            // 31*31
#define THW  3844           // 4*961
#define CCC  256            // attention channels
#define CPP  42             // compare channels per branch
#define S16  16             // Tm*Tq
#define NCMP 15376          // 16*961
#define AGG  1016
#define SROW 976            // padded score-row stride (16B aligned, even)

// ---- A fragment loader: 16x32 f16 layout (ISA 7.12.2) ----
// AVEC: aks==1, row stride even, base 8B-aligned -> 8x float2 loads.
template <bool AVEC>
__device__ __forceinline__ v16h frag_a(const float* __restrict__ pa,
                                       int k0, int hl, unsigned aks) {
  v16h f;
  if (AVEC) {
    const float2* p0 = (const float2*)(pa + k0 + hl * 8);        // v0..3 -> K k0+hl*8 + 2v+p
    const float2* p1 = (const float2*)(pa + k0 + 16 + hl * 8);   // v4..7 -> +16
    float2 t[8];
#pragma unroll
    for (int v = 0; v < 4; ++v) t[v] = p0[v];
#pragma unroll
    for (int v = 0; v < 4; ++v) t[4 + v] = p1[v];
#pragma unroll
    for (int v = 0; v < 8; ++v) {
      f[2 * v]     = (_Float16)t[v].x;
      f[2 * v + 1] = (_Float16)t[v].y;
    }
  } else {
    float t[16];
#pragma unroll
    for (int e = 0; e < 16; ++e) {
      int v = e >> 1, p = e & 1;
      unsigned ka = (unsigned)(k0 + ((v < 4) ? 0 : 16) + hl * 8 + (v & 3) * 2 + p);
      t[e] = pa[ka * aks];                     // unsigned 32-bit index, no sext
    }
#pragma unroll
    for (int e = 0; e < 16; ++e) f[e] = (_Float16)t[e];
  }
  return f;
}

// ---- B fragment loader: 16x32 table layout, e<8 -> K hl*8+e, e>=8 -> K 16+hl*8+(e-8) ----
__device__ __forceinline__ v16h frag_b(const float* __restrict__ pb,
                                       int k0, int hl, unsigned bks) {
  float t[16];
#pragma unroll
  for (int e = 0; e < 16; ++e) t[e] = pb[(unsigned)(k0 + ((e < 8) ? 0 : 16) + hl * 8 + (e & 7)) * bks];
  v16h f;
#pragma unroll
  for (int e = 0; e < 16; ++e) f[e] = (_Float16)t[e];
  return f;
}

// ---- store one 16x16 C tile ----
__device__ __forceinline__ void store_tile(float* __restrict__ Cm, v8f acc,
                                           int mbase, int ncs, int hl,
                                           int M, int N, int cms, int cns,
                                           float alpha, int accumulate) {
  if (ncs >= N) return;
  if (mbase + 16 <= M) {
#pragma unroll
    for (int r = 0; r < 8; ++r) {
      unsigned idx = (unsigned)((mbase + r + 8 * hl) * cms + ncs * cns);
      float v = acc[r] * alpha;
      if (accumulate) v += Cm[idx];
      Cm[idx] = v;
    }
  } else {
#pragma unroll
    for (int r = 0; r < 8; ++r) {
      int mr = mbase + r + 8 * hl;
      if (mr < M) {
        unsigned idx = (unsigned)(mr * cms + ncs * cns);
        float v = acc[r] * alpha;
        if (accumulate) v += Cm[idx];
        Cm[idx] = v;
      }
    }
  }
}

// ---------------------------------------------------------------------------
// Generic WMMA GEMM: C[m,n] (+)= alpha * sum_k A[m,k]*B[k,n]
// One wave computes a 32x32 macro-tile: 2 A frags x 2 B frags -> 4 WMMAs per
// 32-deep K step (16 loads per WMMA). M/N edges handled by clamping load
// rows/cols (garbage lanes never stored). 32-bit unsigned index math.
// ---------------------------------------------------------------------------
__global__ void k_gemm_wmma(const float* __restrict__ A, const float* __restrict__ Bm,
                            float* __restrict__ Cm,
                            int M, int N, int K,
                            int ams, int aks, int bks, int bns, int cms, int cns,
                            long abat, long bbat, long cbat,
                            float alpha, int accumulate, int bnmod) {
  int lane = threadIdx.x & 31;
  int wave = threadIdx.x >> 5;
  int n0 = (blockIdx.x * (blockDim.x >> 5) + wave) * 32;   // two 16-col tiles
  int m0 = blockIdx.y * 32;                                // two 16-row tiles
  long bz = blockIdx.z;
  A  += bz * abat;
  Bm += bz * bbat;
  Cm += bz * cbat;
  if (m0 >= M || n0 >= N) return;

  int hl = lane >> 4;
  int lm = lane & 15;
  int mr0 = m0 + lm;       if (mr0 > M - 1) mr0 = M - 1;
  int mr1 = m0 + 16 + lm;  if (mr1 > M - 1) mr1 = M - 1;
  int nc0 = n0 + lm;       if (nc0 > N - 1) nc0 = N - 1;
  int nc1 = n0 + 16 + lm;  if (nc1 > N - 1) nc1 = N - 1;
  int ncb0 = (bnmod > 0) ? (nc0 % bnmod) : nc0;
  int ncb1 = (bnmod > 0) ? (nc1 % bnmod) : nc1;

  const float* pa0 = A  + (long)mr0 * ams;
  const float* pa1 = A  + (long)mr1 * ams;
  const float* pb0 = Bm + (long)ncb0 * bns;
  const float* pb1 = Bm + (long)ncb1 * bns;
  const bool avec = (aks == 1) && ((ams & 1) == 0) &&
                    ((((unsigned long long)A) & 7ull) == 0ull);   // uniform

  v8f acc00 = {}, acc01 = {}, acc10 = {}, acc11 = {};
  const int kfull = K & ~31;
  // ---- main loops: avec branch hoisted; no guards inside ----
  if (avec) {
    for (int k0 = 0; k0 < kfull; k0 += 32) {
      v16h a0 = frag_a<true>(pa0, k0, hl, (unsigned)aks);
      v16h a1 = frag_a<true>(pa1, k0, hl, (unsigned)aks);
      v16h b0 = frag_b(pb0, k0, hl, (unsigned)bks);
      v16h b1 = frag_b(pb1, k0, hl, (unsigned)bks);
      acc00 = __builtin_amdgcn_wmma_f32_16x16x32_f16(false, a0, false, b0, (short)0, acc00, false, false);
      acc01 = __builtin_amdgcn_wmma_f32_16x16x32_f16(false, a0, false, b1, (short)0, acc01, false, false);
      acc10 = __builtin_amdgcn_wmma_f32_16x16x32_f16(false, a1, false, b0, (short)0, acc10, false, false);
      acc11 = __builtin_amdgcn_wmma_f32_16x16x32_f16(false, a1, false, b1, (short)0, acc11, false, false);
    }
  } else {
    for (int k0 = 0; k0 < kfull; k0 += 32) {
      v16h a0 = frag_a<false>(pa0, k0, hl, (unsigned)aks);
      v16h a1 = frag_a<false>(pa1, k0, hl, (unsigned)aks);
      v16h b0 = frag_b(pb0, k0, hl, (unsigned)bks);
      v16h b1 = frag_b(pb1, k0, hl, (unsigned)bks);
      acc00 = __builtin_amdgcn_wmma_f32_16x16x32_f16(false, a0, false, b0, (short)0, acc00, false, false);
      acc01 = __builtin_amdgcn_wmma_f32_16x16x32_f16(false, a0, false, b1, (short)0, acc01, false, false);
      acc10 = __builtin_amdgcn_wmma_f32_16x16x32_f16(false, a1, false, b0, (short)0, acc10, false, false);
      acc11 = __builtin_amdgcn_wmma_f32_16x16x32_f16(false, a1, false, b1, (short)0, acc11, false, false);
    }
  }
  // ---- K epilogue: clamp addresses, zero values via cndmask ----
  if (kfull < K) {
    int k0 = kfull;
    v16h a0, a1, b0, b1;
#pragma unroll
    for (int e = 0; e < 16; ++e) {
      int v = e >> 1, p = e & 1;
      int ka = k0 + ((v < 4) ? 0 : 16) + hl * 8 + (v & 3) * 2 + p;
      int kb = k0 + ((e < 8) ? 0 : 16) + hl * 8 + (e & 7);
      unsigned kac = (unsigned)((ka < K - 1) ? ka : (K - 1));
      unsigned kbc = (unsigned)((kb < K - 1) ? kb : (K - 1));
      float av0 = pa0[kac * (unsigned)aks];
      float av1 = pa1[kac * (unsigned)aks];
      float bv0 = pb0[kbc * (unsigned)bks];
      float bv1 = pb1[kbc * (unsigned)bks];
      a0[e] = (_Float16)((ka < K) ? av0 : 0.0f);
      a1[e] = (_Float16)((ka < K) ? av1 : 0.0f);
      b0[e] = (_Float16)((kb < K) ? bv0 : 0.0f);
      b1[e] = (_Float16)((kb < K) ? bv1 : 0.0f);
    }
    acc00 = __builtin_amdgcn_wmma_f32_16x16x32_f16(false, a0, false, b0, (short)0, acc00, false, false);
    acc01 = __builtin_amdgcn_wmma_f32_16x16x32_f16(false, a0, false, b1, (short)0, acc01, false, false);
    acc10 = __builtin_amdgcn_wmma_f32_16x16x32_f16(false, a1, false, b0, (short)0, acc10, false, false);
    acc11 = __builtin_amdgcn_wmma_f32_16x16x32_f16(false, a1, false, b1, (short)0, acc11, false, false);
  }

  // ---- writeback: four 16x16 tiles ----
  int ncs0 = n0 + lm, ncs1 = n0 + 16 + lm;
  store_tile(Cm, acc00, m0,      ncs0, hl, M, N, cms, cns, alpha, accumulate);
  store_tile(Cm, acc01, m0,      ncs1, hl, M, N, cms, cns, alpha, accumulate);
  store_tile(Cm, acc10, m0 + 16, ncs0, hl, M, N, cms, cns, alpha, accumulate);
  store_tile(Cm, acc11, m0 + 16, ncs1, hl, M, N, cms, cns, alpha, accumulate);
}

// ---------------- max_pool3d (1,3,3)/(1,2,2) VALID : x(4,512,4,63,63)->xp(...,31,31)
__global__ void k_maxpool(const float* __restrict__ x, float* __restrict__ xp) {
  long i = (long)blockIdx.x * 256 + threadIdx.x;
  const long total = (long)BB * CCH * TT * HP * WP;
  if (i >= total) return;
  int w = (int)(i % WP); long r = i / WP;
  int h = (int)(r % HP); r /= HP;          // r = ((b*512+c)*4+t)
  const float* src = x + (r * 63 + (long)h * 2) * 63 + (long)w * 2;
  float m = src[0];
#pragma unroll
  for (int dh = 0; dh < 3; ++dh)
#pragma unroll
    for (int dw = 0; dw < 3; ++dw)
      m = fmaxf(m, src[dh * 63 + dw]);
  xp[i] = m;
}

// ---------------- row softmax over ncols (in place), rows strided by rstride
__global__ void k_softmax_rows(float* __restrict__ S, int ncols, int rstride) {
  __shared__ float red[128];
  int tid = threadIdx.x;
  float* p = S + (long)blockIdx.x * rstride;
  float m = -3.0e38f;
  for (int j = tid; j < ncols; j += 128) m = fmaxf(m, p[j]);
  red[tid] = m; __syncthreads();
  for (int s = 64; s > 0; s >>= 1) {
    if (tid < s) red[tid] = fmaxf(red[tid], red[tid + s]);
    __syncthreads();
  }
  m = red[0]; __syncthreads();
  float sum = 0.0f;
  for (int j = tid; j < ncols; j += 128) {
    float e = __expf(p[j] - m);
    p[j] = e;                    // stash exp; normalized below
    sum += e;
  }
  red[tid] = sum; __syncthreads();
  for (int s = 64; s > 0; s >>= 1) {
    if (tid < s) red[tid] += red[tid + s];
    __syncthreads();
  }
  float inv = 1.0f / red[0];
  for (int j = tid; j < ncols; j += 128) p[j] *= inv;
}

// ---------------- build compare operands: Rp (R rearranged), Tsub=R-V, Tmul=R*V
// layouts: R[b][tm][q=tq*961+p][c] ; out (b, c, s=tm*4+tq, p)
__global__ void k_prep_compare(const float* __restrict__ R, const float* __restrict__ V,
                               float* __restrict__ Rp, float* __restrict__ Tsub,
                               float* __restrict__ Tmul) {
  long i = (long)blockIdx.x * 256 + threadIdx.x;
  const long total = (long)BB * CCC * S16 * HWP;
  if (i >= total) return;
  int p = (int)(i % HWP); long r = i / HWP;
  int s = (int)(r % S16); r /= S16;
  int c = (int)(r % CCC); int b = (int)(r / CCC);
  int tm = s >> 2, tq = s & 3;
  float rv = R[((long)(b * 4 + tm) * THW + (long)tq * HWP + p) * CCC + c];
  float vv = V[((long)b * CCC + c) * THW + (long)tq * HWP + p];
  Rp[i]   = rv;
  Tsub[i] = rv - vv;
  Tmul[i] = rv * vv;
}

// ---------------- training-mode BN stats: one block per channel
__global__ void k_bnstats(const float* __restrict__ X, float* __restrict__ mean,
                          float* __restrict__ var, long ch_stride, long b_stride,
                          int nb, long ne) {
  __shared__ float s1[256], s2[256];
  int ci = blockIdx.x, tid = threadIdx.x;
  const float* base = X + (long)ci * ch_stride;
  float a = 0.0f, b2 = 0.0f;
  for (int b = 0; b < nb; ++b)
    for (long e = tid; e < ne; e += 256) {
      float v = base[(long)b * b_stride + e];
      a += v; b2 += v * v;
    }
  s1[tid] = a; s2[tid] = b2; __syncthreads();
  for (int s = 128; s > 0; s >>= 1) {
    if (tid < s) { s1[tid] += s1[tid + s]; s2[tid] += s2[tid + s]; }
    __syncthreads();
  }
  if (tid == 0) {
    float cnt = (float)nb * (float)ne;
    float mu = s1[0] / cnt;
    mean[ci] = mu;
    var[ci] = s2[0] / cnt - mu * mu;
  }
}

// ---------------- assemble Y = [mean_tq(R) | V | relu(bn(compare))]  (b,1016,3844)
__global__ void k_buildY(const float* __restrict__ R, const float* __restrict__ V,
                         const float* __restrict__ Ccmp,
                         const float* __restrict__ mean_c, const float* __restrict__ var_c,
                         const float* __restrict__ g_cat, const float* __restrict__ b_cat,
                         const float* __restrict__ g_sub, const float* __restrict__ b_sub,
                         const float* __restrict__ g_mul, const float* __restrict__ b_mul,
                         float* __restrict__ Y) {
  long i = (long)blockIdx.x * 256 + threadIdx.x;
  const long total = (long)BB * AGG * THW;
  if (i >= total) return;
  int n = (int)(i % THW); long r = i / THW;
  int ch = (int)(r % AGG); int b = (int)(r / AGG);
  int t = n / HWP, p = n % HWP;
  float val;
  if (ch < 256) {                       // R.mean over query frames; Y's t = tm
    const float* rb = R + ((long)(b * 4 + t) * THW) * CCC + ch;
    float s = 0.0f;
#pragma unroll
    for (int tq = 0; tq < 4; ++tq) s += rb[((long)tq * HWP + p) * CCC];
    val = s * 0.25f;
  } else if (ch < 512) {
    val = V[((long)b * CCC + (ch - 256)) * THW + n];
  } else {                               // Cfeat reshape: cf = ch126*4 + tm, t = tq
    int cf = ch - 512;
    int ch126 = cf >> 2, tm = cf & 3;
    int branch = ch126 / CPP, chp = ch126 % CPP;
    float v = Ccmp[((long)b * 126 + ch126) * NCMP + (long)(tm * 4 + t) * HWP + p];
    float g, be;
    if      (branch == 0) { g = g_cat[chp]; be = b_cat[chp]; }
    else if (branch == 1) { g = g_sub[chp]; be = b_sub[chp]; }
    else                  { g = g_mul[chp]; be = b_mul[chp]; }
    v = (v - mean_c[ch126]) * rsqrtf(var_c[ch126] + 1e-5f) * g + be;
    val = fmaxf(v, 0.0f);
  }
  Y[i] = val;
}

// ---------------- out = relu(xp + bn(Z))
__global__ void k_final(const float* __restrict__ xp, const float* __restrict__ Z,
                        const float* __restrict__ mz, const float* __restrict__ vz,
                        const float* __restrict__ g, const float* __restrict__ be,
                        float* __restrict__ out) {
  long i = (long)blockIdx.x * 256 + threadIdx.x;
  const long total = (long)BB * CCH * THW;
  if (i >= total) return;
  int c = (int)((i / THW) % CCH);
  float z = Z[i];
  z = (z - mz[c]) * rsqrtf(vz[c] + 1e-5f) * g[c] + be[c];
  out[i] = fmaxf(xp[i] + z, 0.0f);
}

extern "C" void kernel_launch(void* const* d_in, const int* in_sizes, int n_in,
                              void* d_out, int out_size, void* d_ws, size_t ws_size,
                              hipStream_t stream) {
  (void)in_sizes; (void)n_in; (void)out_size; (void)ws_size;
  const float* x     = (const float*)d_in[0];
  const float* Wq    = (const float*)d_in[1];
  const float* Wm    = (const float*)d_in[2];
  const float* Wv    = (const float*)d_in[3];
  const float* Wcat  = (const float*)d_in[4];
  const float* Wsub  = (const float*)d_in[5];
  const float* Wmul  = (const float*)d_in[6];
  const float* g_cat = (const float*)d_in[7];
  const float* b_cat = (const float*)d_in[8];
  const float* g_sub = (const float*)d_in[9];
  const float* b_sub = (const float*)d_in[10];
  const float* g_mul = (const float*)d_in[11];
  const float* b_mul = (const float*)d_in[12];
  const float* Wagg  = (const float*)d_in[13];
  const float* g_agg = (const float*)d_in[14];
  const float* b_agg = (const float*)d_in[15];
  float* out = (float*)d_out;
  float* ws  = (float*)d_ws;

  // workspace layout (floats)
  long o = 0;
  float* xpw   = ws + o; o += (long)BB * CCH * THW;          // 7,872,512
  float* Qw    = ws + o; o += (long)BB * CCC * THW;          // 3,936,256
  float* Mw    = ws + o; o += (long)BB * CCC * THW;
  float* Vw    = ws + o; o += (long)BB * CCC * THW;
  float* Rw    = ws + o; o += (long)BB * 4 * THW * CCC;      // 15,745,024
  float* Sbuf  = ws + o; o += (long)THW * SROW;              // padded score rows
  float* Rp    = ws + o; o += (long)BB * CCC * S16 * HWP;
  float* Tsub  = ws + o; o += (long)BB * CCC * S16 * HWP;
  float* Tmul  = ws + o; o += (long)BB * CCC * S16 * HWP;
  float* Ccmp  = ws + o; o += (long)BB * 126 * NCMP;         // 7,749,504
  float* Yw    = ws + o; o += (long)BB * AGG * THW;          // 15,622,016
  float* Zw    = ws + o; o += (long)BB * CCH * THW;
  float* mean_c = ws + o; o += 126;
  float* var_c  = ws + o; o += 126;
  float* mean_z = ws + o; o += 512;
  float* var_z  = ws + o; o += 512;

  const long PIX = (long)BB * CCH * THW;     // 7,872,512

  // 0) max pool
  k_maxpool<<<(unsigned)((PIX + 255) / 256), 256, 0, stream>>>(x, xpw);

  // 1) Q/M/V projections: (256x512) x (512x3844) per batch
  {
    dim3 g((THW + 127) / 128, (CCC + 31) / 32, BB);
    k_gemm_wmma<<<g, 128, 0, stream>>>(Wq, xpw, Qw, CCC, THW, CCH,
        512, 1, THW, 1, THW, 1, 0, (long)CCH * THW, (long)CCC * THW, 1.0f, 0, 0);
    k_gemm_wmma<<<g, 128, 0, stream>>>(Wm, xpw, Mw, CCC, THW, CCH,
        512, 1, THW, 1, THW, 1, 0, (long)CCH * THW, (long)CCC * THW, 1.0f, 0, 0);
    k_gemm_wmma<<<g, 128, 0, stream>>>(Wv, xpw, Vw, CCC, THW, CCH,
        512, 1, THW, 1, THW, 1, 0, (long)CCH * THW, (long)CCC * THW, 1.0f, 0, 0);
  }

  // 2) attention, chunked per (batch, memory frame): scores -> softmax -> R
  const long QB = (long)CCC * THW;
  for (int b = 0; b < BB; ++b) {
    for (int m = 0; m < TT; ++m) {
      dim3 gs((HWP + 127) / 128, (THW + 31) / 32, 1);
      // S[q,j] = (1/16) * sum_c Q[c,q] * M[c, m*961+j]
      k_gemm_wmma<<<gs, 128, 0, stream>>>(Qw + b * QB, Mw + b * QB + (long)m * HWP, Sbuf,
          THW, HWP, CCC, 1, THW, THW, 1, SROW, 1, 0, 0, 0, 0.0625f, 0, 0);
      k_softmax_rows<<<THW, 128, 0, stream>>>(Sbuf, HWP, SROW);
      dim3 gr((CCC + 127) / 128, (THW + 31) / 32, 1);
      // R[q,c] = sum_j A[q,j] * V[c, m*961+j]
      k_gemm_wmma<<<gr, 128, 0, stream>>>(Sbuf, Vw + b * QB + (long)m * HWP,
          Rw + (long)(b * 4 + m) * THW * CCC,
          THW, CCC, HWP, SROW, 1, 1, THW, CCC, 1, 0, 0, 0, 1.0f, 0, 0);
    }
  }

  // 3) compare operand prep
  {
    long total = (long)BB * CCC * S16 * HWP;
    k_prep_compare<<<(unsigned)((total + 255) / 256), 256, 0, stream>>>(Rw, Vw, Rp, Tsub, Tmul);
  }

  // 4) compare 1x1 convs -> Ccmp (b, branch*42+ch, 16*961)
  {
    dim3 g((NCMP + 127) / 128, (CPP + 31) / 32, BB);
    // Ccat = Wcat[:, :256] @ Rp  +  Wcat[:, 256:] @ V (broadcast over tm via n%3844)
    k_gemm_wmma<<<g, 128, 0, stream>>>(Wcat, Rp, Ccmp, CPP, NCMP, CCC,
        512, 1, NCMP, 1, NCMP, 1, 0, (long)CCC * NCMP, (long)126 * NCMP, 1.0f, 0, 0);
    k_gemm_wmma<<<g, 128, 0, stream>>>(Wcat + 256, Vw, Ccmp, CPP, NCMP, CCC,
        512, 1, THW, 1, NCMP, 1, 0, (long)CCC * THW, (long)126 * NCMP, 1.0f, 1, THW);
    // Csub = Wsub @ (R - V)
    k_gemm_wmma<<<g, 128, 0, stream>>>(Wsub, Tsub, Ccmp + (long)CPP * NCMP, CPP, NCMP, CCC,
        256, 1, NCMP, 1, NCMP, 1, 0, (long)CCC * NCMP, (long)126 * NCMP, 1.0f, 0, 0);
    // Cmul = Wmul @ (R * V)
    k_gemm_wmma<<<g, 128, 0, stream>>>(Wmul, Tmul, Ccmp + (long)(2 * CPP) * NCMP, CPP, NCMP, CCC,
        256, 1, NCMP, 1, NCMP, 1, 0, (long)CCC * NCMP, (long)126 * NCMP, 1.0f, 0, 0);
  }

  // 5) BN stats for the 126 compare channels
  k_bnstats<<<126, 256, 0, stream>>>(Ccmp, mean_c, var_c,
      (long)NCMP, (long)126 * NCMP, BB, (long)NCMP);

  // 6) assemble Y (b, 1016, 3844)
  {
    long total = (long)BB * AGG * THW;
    k_buildY<<<(unsigned)((total + 255) / 256), 256, 0, stream>>>(
        Rw, Vw, Ccmp, mean_c, var_c, g_cat, b_cat, g_sub, b_sub, g_mul, b_mul, Yw);
  }

  // 7) aggregate GEMM: Z = Wagg (512x1016) @ Y (1016x3844) per batch
  {
    dim3 g((THW + 127) / 128, (CCH + 31) / 32, BB);
    k_gemm_wmma<<<g, 128, 0, stream>>>(Wagg, Yw, Zw, CCH, THW, AGG,
        AGG, 1, THW, 1, THW, 1, 0, (long)AGG * THW, (long)CCH * THW, 1.0f, 0, 0);
  }

  // 8) BN stats on Z, then out = relu(xp + bn(Z))
  k_bnstats<<<512, 256, 0, stream>>>(Zw, mean_z, var_z,
      (long)THW, (long)CCH * THW, BB, (long)THW);
  k_final<<<(unsigned)((PIX + 255) / 256), 256, 0, stream>>>(
      xpw, Zw, mean_z, var_z, g_agg, b_agg, out);
}
